// PaliGemmaWithExpertModel_37211596653151
// MI455X (gfx1250) — compile-verified, matrix operand order checked by reading, not run on hardware
//
#include <hip/hip_runtime.h>
#include <hip/hip_bf16.h>
#include <stdint.h>

// ---------------------------------------------------------------------------
// PaliGemma-with-expert forward for MI455X (gfx1250, wave32, WMMA).
// Memory-bound on ~1.02 GB of fp32 weights (~45us @ 23.3 TB/s): weights are
// streamed fp32 -> LDS with global_load_async_to_lds_b128 (double-buffered so
// the DMA of tile k+1 overlaps WMMA on tile k) and down-converted to bf16 in
// registers feeding v_wmma_f32_16x16x32_bf16 (f32 accumulate).
// ---------------------------------------------------------------------------

typedef __attribute__((ext_vector_type(16))) __bf16 v16bf;
typedef __attribute__((ext_vector_type(8)))  float  v8f;

#define B_   4
#define PRE_ 768
#define SUF_ 64
#define T_   832
#define H_   8
#define D_   256
#define HD_  2048
#define BIG_NEG (-2.3819763e+38f)

__device__ __forceinline__ void async_ld_b128(uint32_t ldsOff, const void* g) {
  asm volatile("global_load_async_to_lds_b128 %0, %1, off"
               :: "v"(ldsOff), "v"((uint64_t)(uintptr_t)g)
               : "memory");
}
template <int N>
__device__ __forceinline__ void wait_async_le() {
  asm volatile("s_wait_asynccnt %0" :: "i"(N) : "memory");
}

__device__ __forceinline__ float gelu_tanh_f(float x) {
  float x3 = x * x * x;
  return 0.5f * x * (1.0f + tanhf(0.7978845608028654f * (x + 0.044715f * x3)));
}

// ---------------------------------------------------------------------------
// C[M,N] = A[M,K] @ B + epilogue.   BT=false: B is [K,N] (ldb = row stride).
//                                   BT=true : B is [N,K] (Q*K^T style).
// Batched via grid.z: z -> (zb = z/nh, zh = z%nh), pointer offsets per batch.
// epi: 0 = store, 1 = +R (residual), 2 = C = gelu(C_old) * acc (gate*up fuse).
// Double-buffered async global->LDS pipeline: prefetch tile k+1 while
// computing tile k; s_wait_asynccnt <= TOT retires only the current tile.
// ---------------------------------------------------------------------------
template <int BN, bool BT>
__global__ __launch_bounds__(256) void gemm_wmma(
    const float* __restrict__ A, const float* __restrict__ Bm,
    float* __restrict__ C, const float* __restrict__ R,
    int M, int N, int K, int lda, int ldb, int ldc,
    long long aB, long long bB, long long cB,
    int nh, long long aH, long long bH, long long cH, int epi) {
  (void)M; (void)N;
  constexpr int BROWS = BT ? BN : 32;
  constexpr int BCOLS = BT ? 36 : (BN + 4);
  constexpr int NSUB  = BN / 32;        // 16-wide C sub-tiles per wave
  constexpr int BPF   = (8 * BN) / 256; // async b128 issues per thread for B
  constexpr int TOT   = 2 + BPF;        // per-wave async issues per tile
  constexpr uint32_t ABYTES = 64u * 36u * 4u;
  constexpr uint32_t BBYTES = (uint32_t)BROWS * (uint32_t)BCOLS * 4u;
  __shared__ __align__(16) float Alds[2][64][36];
  __shared__ __align__(16) float Blds[2][BROWS][BCOLS];

  const int z  = blockIdx.z;
  const int zb = z / nh, zh = z - zb * nh;
  const float* Ab = A  + zb * aB + zh * aH;
  const float* Bb = Bm + zb * bB + zh * bH;
  float*       Cb = C  + zb * cB + zh * cH;
  const float* Rb = R  + zb * cB + zh * cH;

  const int m0   = blockIdx.y * 64;
  const int n0   = blockIdx.x * BN;
  const int tid  = threadIdx.x;
  const int lane = tid & 31, wid = tid >> 5;
  const int wm   = (wid & 3) * 16;          // 4 waves along M
  const int wn   = (wid >> 2) * (BN / 2);   // 2 waves along N
  const int ml   = lane & 15, sel = lane >> 4;

  const uint32_t aBase = (uint32_t)(uintptr_t)&Alds[0][0][0];
  const uint32_t bBase = (uint32_t)(uintptr_t)&Blds[0][0][0];

  // Issue one tile's async DMA (A 64x32, B 32xBN or BNx32) into buffer `buf`.
  auto issue = [&](int buf, int kc) {
#pragma unroll
    for (int i = 0; i < 2; ++i) {
      int idx = tid + 256 * i;
      int row = idx >> 3, c4 = idx & 7;
      async_ld_b128(aBase + (uint32_t)buf * ABYTES +
                        (uint32_t)(row * 36 + c4 * 4) * 4u,
                    Ab + (size_t)(m0 + row) * lda + kc + c4 * 4);
    }
    if constexpr (BT) {
#pragma unroll
      for (int i = 0; i < BPF; ++i) {
        int idx = tid + 256 * i;
        int row = idx >> 3, c4 = idx & 7;
        async_ld_b128(bBase + (uint32_t)buf * BBYTES +
                          (uint32_t)(row * BCOLS + c4 * 4) * 4u,
                      Bb + (size_t)(n0 + row) * ldb + kc + c4 * 4);
      }
    } else {
#pragma unroll
      for (int i = 0; i < BPF; ++i) {
        int idx = tid + 256 * i;
        int row = idx >> 5, c4 = idx & 31;
        async_ld_b128(bBase + (uint32_t)buf * BBYTES +
                          (uint32_t)(row * BCOLS + c4 * 4) * 4u,
                      Bb + (size_t)(kc + row) * ldb + n0 + c4 * 4);
      }
    }
  };

  v8f acc[NSUB];
#pragma unroll
  for (int t = 0; t < NSUB; ++t)
#pragma unroll
    for (int r = 0; r < 8; ++r) acc[t][r] = 0.0f;

  const int kSteps = K >> 5;
  issue(0, 0);
  for (int ks = 0; ks < kSteps; ++ks) {
    const int cur = ks & 1;
    if (ks + 1 < kSteps) {
      issue(cur ^ 1, (ks + 1) << 5);   // overlap DMA of next tile
      wait_async_le<TOT>();            // current tile retired (in-order)
    } else {
      wait_async_le<0>();
    }
    __syncthreads();                   // all waves' DMAs visible in LDS

    // ---- A fragment: ISA 16-bit A 16x32 layout, fp32->bf16 in regs ----
    v16bf af;
#pragma unroll
    for (int j = 0; j < 8; ++j) {
      int k = (j >> 2) * 16 + sel * 8 + (j & 3) * 2;
      const float2 p = *(const float2*)&Alds[cur][wm + ml][k];
      af[2 * j]     = (__bf16)p.x;
      af[2 * j + 1] = (__bf16)p.y;
    }
#pragma unroll
    for (int tn = 0; tn < NSUB; ++tn) {
      const int gn = wn + tn * 16 + ml;
      v16bf bfv;
#pragma unroll
      for (int j = 0; j < 8; ++j) {
        int k = (j >> 2) * 16 + sel * 8 + (j & 3) * 2;
        float b0, b1;
        if constexpr (BT) {
          const float2 p = *(const float2*)&Blds[cur][gn][k];
          b0 = p.x; b1 = p.y;
        } else {
          b0 = Blds[cur][k][gn];
          b1 = Blds[cur][k + 1][gn];
        }
        bfv[2 * j]     = (__bf16)b0;
        bfv[2 * j + 1] = (__bf16)b1;
      }
      acc[tn] = __builtin_amdgcn_wmma_f32_16x16x32_bf16(
          false, af, false, bfv, (short)0, acc[tn], false, false);
    }
    __syncthreads();  // done reading `cur` before next prefetch overwrites it
  }

  // ---- epilogue: C layout VGPR r -> (M=wm+sel*8+r, N=lane&15) ----
#pragma unroll
  for (int tn = 0; tn < NSUB; ++tn) {
#pragma unroll
    for (int r = 0; r < 8; ++r) {
      const int row = m0 + wm + sel * 8 + r;
      const int col = n0 + wn + tn * 16 + ml;
      size_t off = (size_t)row * ldc + col;
      float v = acc[tn][r];
      if (epi == 1)      v += Rb[off];
      else if (epi == 2) v  = gelu_tanh_f(Cb[off]) * v;
      Cb[off] = v;
    }
  }
}

// ---------------------------------------------------------------------------
__global__ __launch_bounds__(256) void rmsnorm_kernel(
    const float* __restrict__ x, const float* __restrict__ w,
    float* __restrict__ out, int d) {
  const int row = blockIdx.x;
  const float* xr = x + (size_t)row * d;
  float* orow = out + (size_t)row * d;
  __shared__ float red[256];
  float ss = 0.f;
  for (int j = threadIdx.x; j < d; j += 256) { float v = xr[j]; ss += v * v; }
  red[threadIdx.x] = ss; __syncthreads();
  for (int s = 128; s > 0; s >>= 1) {
    if (threadIdx.x < s) red[threadIdx.x] += red[threadIdx.x + s];
    __syncthreads();
  }
  const float inv = rsqrtf(red[0] / (float)d + 1e-6f);
  for (int j = threadIdx.x; j < d; j += 256)
    orow[j] = xr[j] * inv * (1.0f + w[j]);
}

__global__ void rope_kernel(float* __restrict__ x, const int* __restrict__ pos,
                            int nheads) {
  const int idx = blockIdx.x * blockDim.x + threadIdx.x;
  const int total = B_ * T_ * nheads * 128;
  if (idx >= total) return;
  const int i = idx & 127;
  const int h = (idx >> 7) % nheads;
  const int t = ((idx >> 7) / nheads) % T_;
  const int b = idx / (128 * nheads * T_);
  const float p  = (float)pos[b * T_ + t];
  const float ts = __powf(10000.0f, (float)i * (1.0f / 128.0f));
  float sn, cs; __sincosf(p / ts, &sn, &cs);
  const size_t base = ((size_t)(b * T_ + t) * nheads + h) * 256;
  const float x1 = x[base + i], x2 = x[base + 128 + i];
  x[base + i]       = x1 * cs - x2 * sn;
  x[base + 128 + i] = x2 * cs + x1 * sn;
}

__global__ __launch_bounds__(256) void softmax_kernel(
    float* __restrict__ sc, const unsigned char* __restrict__ mask, float scale) {
  const int row = blockIdx.x;  // b*H*T + h*T + q
  const int q = row % T_;
  const int b = row / (H_ * T_);
  float* p = sc + (size_t)row * T_;
  const unsigned char* mrow = mask + ((size_t)b * T_ + q) * T_;
  __shared__ float red[256];
  float lv[4];
  float mx = -3.4e38f;
#pragma unroll
  for (int it = 0; it < 4; ++it) {
    int j = threadIdx.x + it * 256;
    float v = -3.4e38f;
    if (j < T_) { v = p[j] * scale; if (!mrow[j]) v = BIG_NEG; }
    lv[it] = v; mx = fmaxf(mx, v);
  }
  red[threadIdx.x] = mx; __syncthreads();
  for (int s = 128; s > 0; s >>= 1) {
    if (threadIdx.x < s) red[threadIdx.x] = fmaxf(red[threadIdx.x], red[threadIdx.x + s]);
    __syncthreads();
  }
  mx = red[0]; __syncthreads();
  float sum = 0.f;
#pragma unroll
  for (int it = 0; it < 4; ++it) {
    int j = threadIdx.x + it * 256;
    if (j < T_) { lv[it] = __expf(lv[it] - mx); sum += lv[it]; }
  }
  red[threadIdx.x] = sum; __syncthreads();
  for (int s = 128; s > 0; s >>= 1) {
    if (threadIdx.x < s) red[threadIdx.x] += red[threadIdx.x + s];
    __syncthreads();
  }
  const float inv = 1.0f / red[0];
#pragma unroll
  for (int it = 0; it < 4; ++it) {
    int j = threadIdx.x + it * 256;
    if (j < T_) p[j] = lv[it] * inv;
  }
}

// ---------------------------------------------------------------------------
extern "C" void kernel_launch(void* const* d_in, const int* in_sizes, int n_in,
                              void* d_out, int out_size, void* d_ws,
                              size_t ws_size, hipStream_t stream) {
  (void)in_sizes; (void)n_in; (void)out_size; (void)ws_size;
  const float* prefix = (const float*)d_in[0];
  const float* suffix = (const float*)d_in[1];
  const int* posid = (const int*)d_in[2];
  const unsigned char* amask = (const unsigned char*)d_in[3];

  // Params flattened in sorted-key pytree order:
  //  4: exp.final_norm, 5..13 exp.L0{down,gate,in_norm,k,o,post,q,up,v},
  //  14..22 exp.L1, 23: vlm.final_norm, 24..32 vlm.L0, 33..41 vlm.L1
  auto P = [&](int idx) { return (const float*)d_in[idx]; };
  const int EXPFN = 4, VLMFN = 23;
  auto lbase = [&](int s, int li) { return (s == 0 ? 24 : 5) + li * 9; };

  float* ws = (float*)d_ws;
  size_t o = 0;
  float* xv = ws + o; o += (size_t)B_ * PRE_ * 2048;   // vlm stream
  float* xe = ws + o; o += (size_t)B_ * SUF_ * 1024;   // exp stream
  float* hb = ws + o; o += (size_t)B_ * PRE_ * 2048;   // normed activations
  float* qb = ws + o; o += (size_t)B_ * T_ * HD_;      // (B,T,H,256)
  float* kb = ws + o; o += (size_t)B_ * T_ * D_;       // (B,T,256)
  float* vb = ws + o; o += (size_t)B_ * T_ * D_;
  float* sc = ws + o; o += (size_t)B_ * H_ * T_ * T_;  // scores/probs
  float* at = ws + o; o += (size_t)B_ * T_ * HD_;      // attention out
  float* gb = ws + o; o += (size_t)1024 * 16384;       // MLP chunk buffer
  // total = 64M floats = 256 MB

  hipMemcpyAsync(xv, prefix, sizeof(float) * (size_t)B_ * PRE_ * 2048,
                 hipMemcpyDeviceToDevice, stream);
  hipMemcpyAsync(xe, suffix, sizeof(float) * (size_t)B_ * SUF_ * 1024,
                 hipMemcpyDeviceToDevice, stream);

  auto NN = [&](const float* A, const float* W, float* C, const float* R,
                int M, int N, int K, int lda, int ldb, int ldc,
                long long aB, long long bB, long long cB, int batB, int nh,
                long long aH, long long bH, long long cH, int epi) {
    dim3 g(N / 128, M / 64, batB * nh);
    gemm_wmma<128, false><<<g, 256, 0, stream>>>(
        A, W, C, R ? R : C, M, N, K, lda, ldb, ldc, aB, bB, cB, nh, aH, bH, cH, epi);
  };

  for (int li = 0; li < 2; ++li) {
    // ---- QKV projections, both streams into concatenated token axis ----
    for (int s = 0; s < 2; ++s) {
      const int L = s ? SUF_ : PRE_;
      const int d = s ? 1024 : 2048;
      float* x = s ? xe : xv;
      const int toff = s ? PRE_ : 0;
      const int lb = lbase(s, li);
      rmsnorm_kernel<<<B_ * L, 256, 0, stream>>>(x, P(lb + 2), hb, d);
      NN(hb, P(lb + 6), qb + (size_t)toff * HD_, nullptr, L, HD_, d, d, HD_, HD_,
         (long long)L * d, 0, (long long)T_ * HD_, B_, 1, 0, 0, 0, 0);
      NN(hb, P(lb + 3), kb + (size_t)toff * D_, nullptr, L, D_, d, d, D_, D_,
         (long long)L * d, 0, (long long)T_ * D_, B_, 1, 0, 0, 0, 0);
      NN(hb, P(lb + 8), vb + (size_t)toff * D_, nullptr, L, D_, d, d, D_, D_,
         (long long)L * d, 0, (long long)T_ * D_, B_, 1, 0, 0, 0, 0);
    }
    // ---- RoPE ----
    {
      int tq = B_ * T_ * H_ * 128;
      rope_kernel<<<(tq + 255) / 256, 256, 0, stream>>>(qb, posid, H_);
      int tk = B_ * T_ * 1 * 128;
      rope_kernel<<<(tk + 255) / 256, 256, 0, stream>>>(kb, posid, 1);
    }
    // ---- scores = Q @ K^T  (WMMA, B transposed), batch = B*H ----
    {
      dim3 g(T_ / 64, T_ / 64, B_ * H_);
      gemm_wmma<64, true><<<g, 256, 0, stream>>>(
          qb, kb, sc, sc, T_, T_, D_, HD_, D_, T_,
          (long long)T_ * HD_, (long long)T_ * D_, (long long)H_ * T_ * T_,
          H_, (long long)D_, 0, (long long)T_ * T_, 0);
    }
    softmax_kernel<<<B_ * H_ * T_, 256, 0, stream>>>(sc, amask, 0.0625f);
    // ---- att = P @ V (WMMA), batch = B*H ----
    NN(sc, vb, at, nullptr, T_, D_, T_, T_, D_, HD_,
       (long long)H_ * T_ * T_, (long long)T_ * D_, (long long)T_ * HD_, B_,
       H_, (long long)T_ * T_, 0, (long long)D_, 0);
    // ---- o-proj + residual, then MLP (chunked, gate*up fused epilogue) ----
    for (int s = 0; s < 2; ++s) {
      const int L = s ? SUF_ : PRE_;
      const int d = s ? 1024 : 2048;
      const int m = s ? 4096 : 16384;
      float* x = s ? xe : xv;
      const int toff = s ? PRE_ : 0;
      const int lb = lbase(s, li);
      NN(at + (size_t)toff * HD_, P(lb + 4), x, x, L, d, HD_, HD_, d, d,
         (long long)T_ * HD_, 0, (long long)L * d, B_, 1, 0, 0, 0, 1);
      rmsnorm_kernel<<<B_ * L, 256, 0, stream>>>(x, P(lb + 5), hb, d);
      const int rowsTot = B_ * L;
      const int CH = rowsTot < 1024 ? rowsTot : 1024;
      for (int c0 = 0; c0 < rowsTot; c0 += CH) {
        NN(hb + (size_t)c0 * d, P(lb + 1), gb, nullptr, CH, m, d, d, m, m,
           0, 0, 0, 1, 1, 0, 0, 0, 0);                       // g = h@gate
        NN(hb + (size_t)c0 * d, P(lb + 7), gb, nullptr, CH, m, d, d, m, m,
           0, 0, 0, 1, 1, 0, 0, 0, 2);                       // g = gelu(g)*(h@up)
        NN(gb, P(lb + 0), x + (size_t)c0 * d, x + (size_t)c0 * d, CH, d, m,
           m, d, d, 0, 0, 0, 1, 1, 0, 0, 0, 1);              // x += g@down
      }
    }
  }

  float* out = (float*)d_out;
  rmsnorm_kernel<<<B_ * PRE_, 256, 0, stream>>>(xv, P(VLMFN), out, 2048);
  rmsnorm_kernel<<<B_ * SUF_, 256, 0, stream>>>(
      xe, P(EXPFN), out + (size_t)B_ * PRE_ * 2048, 1024);
}
